// SAE_31430570672806
// MI455X (gfx1250) — matrix operand order repaired and checked by reading, not compiled, not run on hardware
//
#include <hip/hip_runtime.h>

// ---------------- problem constants ----------------
#define HIDDEN 4096
#define LATENT 32768
#define NTOK   8192
#define TOPK   64

// ---------------- GEMM tiling ----------------
#define BM 256          // rows (tokens) per workgroup
#define BN 128          // cols (latents) per workgroup
#define KT 32           // K slice per iteration == WMMA K
#define SA 40           // LDS row stride (bf16 elems): 80B = 64B row + 16B TDM pad
#define SB 40
#define NIT (HIDDEN / KT)   // 128 K iterations

typedef __attribute__((ext_vector_type(16))) __bf16 v16bf;
typedef __attribute__((ext_vector_type(8)))  __bf16 v8bf;
typedef __attribute__((ext_vector_type(8)))  float  v8f;

#if __has_builtin(__builtin_amdgcn_tensor_load_to_lds)
#define HAVE_TDM 1
typedef __attribute__((ext_vector_type(4))) unsigned uint32x4;
typedef __attribute__((ext_vector_type(8))) int      int32x8;
typedef __attribute__((ext_vector_type(4))) int      int32x4;
#else
#define HAVE_TDM 0
#endif

// =====================================================================
// Kernel 0: one-time fp32 -> bf16 (RNE) pack into workspace.
// =====================================================================
__global__ __launch_bounds__(256)
void sae_pack_bf16(const float4* __restrict__ src, uint2* __restrict__ dst) {
    const size_t i = (size_t)blockIdx.x * 256 + threadIdx.x;
    const float4 v = src[i];
    unsigned a = __builtin_bit_cast(unsigned, v.x);
    unsigned b = __builtin_bit_cast(unsigned, v.y);
    unsigned c = __builtin_bit_cast(unsigned, v.z);
    unsigned d = __builtin_bit_cast(unsigned, v.w);
    a += 0x7FFFu + ((a >> 16) & 1u);
    b += 0x7FFFu + ((b >> 16) & 1u);
    c += 0x7FFFu + ((c >> 16) & 1u);
    d += 0x7FFFu + ((d >> 16) & 1u);
    uint2 o;
    o.x = __builtin_amdgcn_perm(b, a, 0x07060302u);  // {b.hi16, a.hi16}
    o.y = __builtin_amdgcn_perm(d, c, 0x07060302u);
    dst[i] = o;
}

// =====================================================================
// TDM: one-descriptor 2D tile DMA global(bf16,row-major) -> LDS, with
// hardware padding to produce the 80B bank-skewed LDS row stride.
// D# packing per CDNA5 ISA.
// This toolchain exposes the 6-arg builtin:
//   (uint32x4 g0, int32x8 g1, int32x4 g2, int32x4 g3, int32x8 extra, i32 cpol)
// =====================================================================
#if HAVE_TDM
__device__ __forceinline__ void tdm_load_2d(unsigned lds_byte_off, const void* gaddr,
                                            unsigned tdim0, unsigned tdim1,
                                            unsigned tile0, unsigned tile1,
                                            unsigned stride0) {
    const unsigned long long ga = (unsigned long long)(uintptr_t)gaddr;
    const uint32x4 g0 = {
        1u,                                               // count=1, user mode
        lds_byte_off,                                     // lds_addr
        (unsigned)(ga & 0xFFFFFFFFu),                     // global_addr[31:0]
        (unsigned)((ga >> 32) & 0x01FFFFFFu) | (2u << 30) // global_addr[56:32] | type=2
    };
    const int32x8 g1 = {
        (int)(0x00010000u          // data_size = 2B (bf16)
              | 0x00100000u        // pad_enable
              | (3u << 22)         // pad_interval: 16 dwords (=64B tile row)
              | (3u << 25)),       // pad_amount:    4 dwords (=16B skew)
        (int)(tdim0 << 16),                        // tensor_dim0[15:0]
        (int)((tdim0 >> 16) | (tdim1 << 16)),      // tensor_dim0[31:16] | tensor_dim1[15:0]
        (int)((tdim1 >> 16) | (tile0 << 16)),      // tensor_dim1[31:16] | tile_dim0
        (int)(tile1 & 0xFFFFu),                    // tile_dim1 (tile_dim2=0)
        (int)stride0,                              // tensor_dim0_stride[31:0]
        0, 0
    };
    const int32x4 gz4 = {0, 0, 0, 0};
    const int32x8 gz8 = {0, 0, 0, 0, 0, 0, 0, 0};
    __builtin_amdgcn_tensor_load_to_lds(g0, g1, gz4, gz4, gz8, 0);
}
#endif

// per-lane async fallback (if TDM builtin unavailable)
__device__ __forceinline__ void async_b128(unsigned lds_off, unsigned long long gaddr) {
    asm volatile("global_load_async_to_lds_b128 %0, %1, off"
                 :: "v"(lds_off), "v"(gaddr) : "memory");
}

// =====================================================================
// Kernel 1: encoder GEMM  z[n,l] = sum_h x[n,h]*We[l,h] + be[l]
// bf16 WMMA 16x16x32, fp32 accumulate, double-buffered TDM staging.
// 512 threads = 16 waves in 4(M)x4(N); each wave: 64x32 = 4x2 WMMA tiles.
// =====================================================================
__global__ __launch_bounds__(512)
void sae_enc_gemm(const unsigned short* __restrict__ xb,
                  const unsigned short* __restrict__ Web,
                  const float* __restrict__ be, float* __restrict__ z) {
    __shared__ unsigned short As[2][BM * SA];
    __shared__ unsigned short Bs[2][BN * SB];

    const int tid  = threadIdx.x;
    const int wave = tid >> 5;
    const int lane = tid & 31;
    const int half = lane >> 4;
    const int l16  = lane & 15;

    const int m0 = blockIdx.y * BM;
    const int n0 = blockIdx.x * BN;
    const int waveM = (wave & 3) * 64;      // 4 waves along M
    const int waveN = (wave >> 2) * 32;     // 4 waves along N

    const unsigned asBase[2] = { (unsigned)(uintptr_t)&As[0][0],
                                 (unsigned)(uintptr_t)&As[1][0] };
    const unsigned bsBase[2] = { (unsigned)(uintptr_t)&Bs[0][0],
                                 (unsigned)(uintptr_t)&Bs[1][0] };

    auto issue = [&](int buf, int k0) {
#if HAVE_TDM
        if (wave == 0) {   // one wave drives the DMA engine (TENSORcnt on wave 0)
            tdm_load_2d(asBase[buf], xb  + (size_t)m0 * HIDDEN + k0,
                        HIDDEN, NTOK,   KT, BM, HIDDEN);
            tdm_load_2d(bsBase[buf], Web + (size_t)n0 * HIDDEN + k0,
                        HIDDEN, LATENT, KT, BN, HIDDEN);
        }
#else
        // A tile: 256 rows x 64B = 1024 x 16B chunks (2/thread); B: 512 (1/thread)
#pragma unroll
        for (int i = 0; i < 2; ++i) {
            const int g = i * 512 + tid;
            const int r = g >> 2;
            const int c = (g & 3) * 16;
            async_b128(asBase[buf] + (unsigned)(r * (SA * 2)) + c,
                       (unsigned long long)(uintptr_t)(xb + (size_t)(m0 + r) * HIDDEN + k0) + c);
        }
        {
            const int r = tid >> 2;
            const int c = (tid & 3) * 16;
            async_b128(bsBase[buf] + (unsigned)(r * (SB * 2)) + c,
                       (unsigned long long)(uintptr_t)(Web + (size_t)(n0 + r) * HIDDEN + k0) + c);
        }
#endif
    };

    v8f acc[4][2];
#pragma unroll
    for (int mf = 0; mf < 4; ++mf)
#pragma unroll
        for (int cf = 0; cf < 2; ++cf)
#pragma unroll
            for (int r = 0; r < 8; ++r) acc[mf][cf][r] = 0.0f;

    issue(0, 0);

#pragma unroll 1
    for (int k = 0; k < NIT; ++k) {
        const int cur = k & 1;
        if (k + 1 < NIT) {
            issue(1 - cur, (k + 1) * KT);            // prefetch next slice
#if HAVE_TDM
            if (wave == 0) __builtin_amdgcn_s_wait_tensorcnt(2);  // oldest pair done
#else
            asm volatile("s_wait_asynccnt 0x3" ::: "memory");
#endif
        } else {
#if HAVE_TDM
            if (wave == 0) __builtin_amdgcn_s_wait_tensorcnt(0);
#else
            asm volatile("s_wait_asynccnt 0x0" ::: "memory");
#endif
        }
        __syncthreads();   // publish DMA'd LDS data to all waves

        const unsigned short* Ab = As[cur];
        const unsigned short* Bb = Bs[cur];

        v16bf bfrag[2];
#pragma unroll
        for (int cf = 0; cf < 2; ++cf) {
            const unsigned short* bp = &Bb[(waveN + cf * 16 + l16) * SB + half * 8];
            v8bf lo = *(const v8bf*)bp;
            v8bf hi = *(const v8bf*)(bp + 16);
            bfrag[cf] = __builtin_shufflevector(lo, hi, 0,1,2,3,4,5,6,7,8,9,10,11,12,13,14,15);
        }
#pragma unroll
        for (int mf = 0; mf < 4; ++mf) {
            const unsigned short* ap = &Ab[(waveM + mf * 16 + l16) * SA + half * 8];
            v8bf lo = *(const v8bf*)ap;          // K = half*8 .. +7
            v8bf hi = *(const v8bf*)(ap + 16);   // K = 16+half*8 .. +7
            v16bf afrag = __builtin_shufflevector(lo, hi, 0,1,2,3,4,5,6,7,8,9,10,11,12,13,14,15);
#pragma unroll
            for (int cf = 0; cf < 2; ++cf) {
                acc[mf][cf] = __builtin_amdgcn_wmma_f32_16x16x32_bf16(
                    false, afrag, false, bfrag[cf], (short)0, acc[mf][cf], false, false);
            }
        }
        __syncthreads();   // reads done before this buffer is overwritten
    }

    // ---- epilogue: bias + store dense z tile ----
#pragma unroll
    for (int cf = 0; cf < 2; ++cf) {
        const int col = n0 + waveN + cf * 16 + l16;
        const float bias = be[col];
#pragma unroll
        for (int mf = 0; mf < 4; ++mf) {
#pragma unroll
            for (int r = 0; r < 8; ++r) {
                const int row = m0 + waveM + mf * 16 + half * 8 + r;
                z[(size_t)row * LATENT + col] = acc[mf][cf][r] + bias;
            }
        }
    }
}

// =====================================================================
// Kernel 2: transpose W_dec [HIDDEN][LATENT] -> WdT [LATENT][HIDDEN]
// =====================================================================
__global__ __launch_bounds__(256)
void sae_wdec_transpose(const float* __restrict__ Wd, float* __restrict__ WdT) {
    __shared__ float t[32][33];
    const int l0 = blockIdx.x * 32;
    const int h0 = blockIdx.y * 32;
    const int tx = threadIdx.x & 31;
    const int ty = threadIdx.x >> 5;
#pragma unroll
    for (int i = 0; i < 4; ++i) {
        const int h = ty * 4 + i;
        t[h][tx] = Wd[(size_t)(h0 + h) * LATENT + l0 + tx];
    }
    __syncthreads();
#pragma unroll
    for (int i = 0; i < 4; ++i) {
        const int l = ty * 4 + i;
        WdT[(size_t)(l0 + l) * HIDDEN + h0 + tx] = t[tx][l];
    }
}

// =====================================================================
// Kernel 3: per-token top-64 (row staged in 128KB of the 320KB LDS),
// in-place sparsify of z, sparse decode x_hat = sum_j v_j * WdT[idx_j,:]
// =====================================================================
__global__ __launch_bounds__(256)
void sae_topk_decode(const float* __restrict__ WdT, float* __restrict__ z,
                     float* __restrict__ xhat) {
    extern __shared__ float sm[];
    float* zrow = sm;                         // LATENT floats
    float* redV = sm + LATENT;                // 256
    int*   redI = (int*)(redV + 256);         // 256
    float* topv = (float*)(redI + 256);       // 64
    int*   topi = (int*)(topv + 64);          // 64

    const int tid = threadIdx.x;
    const int row = blockIdx.x;
    float* zg = z + (size_t)row * LATENT;

    float4* z4  = (float4*)zrow;
    float4* zg4 = (float4*)zg;
#pragma unroll 4
    for (int s = 0; s < LATENT / 4 / 256; ++s) z4[s * 256 + tid] = zg4[s * 256 + tid];
    __syncthreads();

    for (int it = 0; it < TOPK; ++it) {
        float best = -__builtin_inff();
        int   bidx = 0;
#pragma unroll 8
        for (int s = 0; s < LATENT / 256; ++s) {
            const int i = s * 256 + tid;
            const float v = zrow[i];
            if (v > best) { best = v; bidx = i; }
        }
        redV[tid] = best; redI[tid] = bidx;
        __syncthreads();
        for (int off = 128; off > 0; off >>= 1) {
            if (tid < off && redV[tid + off] > redV[tid]) {
                redV[tid] = redV[tid + off]; redI[tid] = redI[tid + off];
            }
            __syncthreads();
        }
        if (tid == 0) {
            topv[it] = redV[0]; topi[it] = redI[0];
            zrow[redI[0]] = -__builtin_inff();
        }
        __syncthreads();
    }

    // rebuild sparse row in LDS, write back coalesced
#pragma unroll 8
    for (int s = 0; s < LATENT / 256; ++s) zrow[s * 256 + tid] = 0.0f;
    __syncthreads();
    if (tid < TOPK) zrow[topi[tid]] = topv[tid];
    __syncthreads();
#pragma unroll 4
    for (int s = 0; s < LATENT / 4 / 256; ++s) zg4[s * 256 + tid] = z4[s * 256 + tid];

    // sparse decoder
    float4 acc[4];
#pragma unroll
    for (int s = 0; s < 4; ++s) acc[s] = make_float4(0.f, 0.f, 0.f, 0.f);
    for (int j = 0; j < TOPK; ++j) {
        const float v = topv[j];
        const float4* wr = (const float4*)(WdT + (size_t)topi[j] * HIDDEN);
#pragma unroll
        for (int s = 0; s < 4; ++s) {
            const float4 w = wr[s * 256 + tid];
            acc[s].x += v * w.x; acc[s].y += v * w.y;
            acc[s].z += v * w.z; acc[s].w += v * w.w;
        }
    }
    float4* xo = (float4*)(xhat + (size_t)row * HIDDEN);
#pragma unroll
    for (int s = 0; s < 4; ++s) xo[s * 256 + tid] = acc[s];
}

// =====================================================================
extern "C" void kernel_launch(void* const* d_in, const int* in_sizes, int n_in,
                              void* d_out, int out_size, void* d_ws, size_t ws_size,
                              hipStream_t stream) {
    const float* x  = (const float*)d_in[0];   // [NTOK, HIDDEN]
    const float* We = (const float*)d_in[1];   // [LATENT, HIDDEN]
    const float* be = (const float*)d_in[2];   // [LATENT]
    const float* Wd = (const float*)d_in[3];   // [HIDDEN, LATENT]

    float* xhat = (float*)d_out;
    float* zsp  = (float*)d_out + (size_t)NTOK * HIDDEN;

    // workspace: [ WdT fp32 512MiB | x_bf16 64MiB | We_bf16 256MiB ]
    float*          WdT = (float*)d_ws;
    unsigned short* xb  = (unsigned short*)((char*)d_ws + (size_t)LATENT * HIDDEN * 4);
    unsigned short* Web = xb + (size_t)NTOK * HIDDEN;

    // 0) one-time bf16 packs
    sae_pack_bf16<<<(NTOK * HIDDEN / 4) / 256, 256, 0, stream>>>(
        (const float4*)x, (uint2*)xb);
    sae_pack_bf16<<<(int)(((size_t)LATENT * HIDDEN / 4) / 256), 256, 0, stream>>>(
        (const float4*)We, (uint2*)Web);

    // 1) transpose decoder weights
    sae_wdec_transpose<<<dim3(LATENT / 32, HIDDEN / 32), 256, 0, stream>>>(Wd, WdT);

    // 2) dense encoder GEMM -> z (into z_sparse region, sparsified in place later)
    sae_enc_gemm<<<dim3(LATENT / BN, NTOK / BM), 512, 0, stream>>>(xb, Web, be, zsp);

    // 3) top-k + sparsify + sparse decode
    const size_t smem = (size_t)(LATENT + 256 + 256 + 64 + 64) * 4;
    sae_topk_decode<<<NTOK, 256, smem, stream>>>(WdT, zsp, xhat);
}
